// SAMN_TRUST_9096740733368
// MI455X (gfx1250) — matrix operand order repaired and checked by reading, not compiled, not run on hardware
//
#include <hip/hip_runtime.h>
#include <stdint.h>

// ---------------------------------------------------------------------------
// SAMN trust pipeline for MI455X (gfx1250, wave32, WMMA bf16 16x16x32)
//   k0: convert Key/Mem/WA to bf16 WMMA-B fragment layout (L2-resident tables)
//   k1: per-batch block: gather uid/fe (b128, mask-by-multiply), L2-normalize,
//       store cross (bf16), att_key = cross@Key  (WMMA)
//   k2: softmax-over-batch stats: per (f,m) max & sum-exp over 4096 batch rows
//   k3: per-batch block: TDM/async stage att_key -> LDS, att_mem (bf16),
//       f1=att_mem@Mem (WMMA, fused f2=f1*fe), h=relu(f2@WA+BA) (WMMA, fused
//       j=exp(h@U)), w, friend, 2-way attn, score
// ---------------------------------------------------------------------------

typedef __attribute__((ext_vector_type(16))) __bf16 v16bf;
typedef __attribute__((ext_vector_type(8)))  float  v8f;
typedef __attribute__((ext_vector_type(4)))  unsigned int v4u;
typedef __attribute__((ext_vector_type(8)))  int v8i;
typedef __attribute__((ext_vector_type(4)))  int v4i;
typedef unsigned short u16;

#define GAS __attribute__((address_space(1)))
#define LAS __attribute__((address_space(3)))

#if __has_builtin(__builtin_amdgcn_tensor_load_to_lds) && __has_builtin(__builtin_amdgcn_s_wait_tensorcnt)
#define HAVE_TDM 1
#else
#define HAVE_TDM 0
#endif
#if __has_builtin(__builtin_amdgcn_global_load_async_to_lds_b128) && __has_builtin(__builtin_amdgcn_s_wait_asynccnt)
#define HAVE_ASYNC 1
#else
#define HAVE_ASYNC 0
#endif

#define USER_NUMC 500000
#define DD   256
#define MM   32
#define NFF  64
#define BB   4096

// workspace layout (bytes)
#define OFF_ATT   0u                       // att_key  f32 [B][64][32]   32 MB
#define OFF_MAX   33554432u                // max      f32 [64][32]
#define OFF_SUM   33562624u                // sumexp   f32 [64][32]
#define OFF_FKEY  33570816u                // Key frags bf16 [8192]
#define OFF_FMEM  33587200u                // Mem frags bf16 [8192]
#define OFF_FWA   33603584u                // WA  frags bf16 [65536]

__device__ __forceinline__ float bf2f(u16 u) {
  unsigned int x = ((unsigned int)u) << 16;
  float f; __builtin_memcpy(&f, &x, 4); return f;
}
__device__ __forceinline__ u16 f2bf(float f) {
  __bf16 h = (__bf16)f;
  u16 u; __builtin_memcpy(&u, &h, 2); return u;
}
__device__ __forceinline__ unsigned int pk2(float a, float b) {
  union { __bf16 h[2]; unsigned int u; } c;
  c.h[0] = (__bf16)a; c.h[1] = (__bf16)b; return c.u;
}

__device__ __forceinline__ float blockReduceSum(float v, float* scratch) {
#pragma unroll
  for (int off = 16; off > 0; off >>= 1) v += __shfl_xor(v, off, 32);
  int wid = threadIdx.x >> 5;
  if ((threadIdx.x & 31) == 0) scratch[wid] = v;
  __syncthreads();
  if (threadIdx.x < 8) {
    v = scratch[threadIdx.x];
#pragma unroll
    for (int off = 4; off > 0; off >>= 1) v += __shfl_xor(v, off, 32);
    if (threadIdx.x == 0) scratch[0] = v;
  }
  __syncthreads();
  v = scratch[0];
  __syncthreads();
  return v;
}

// ---------------------------------------------------------------------------
// k0: pack B-operands into per-lane-contiguous bf16 fragment tables.
// Fragment element order: (((nt*ktiles + kb)*32 + lane)*16 + e), where the
// source element is  src[(kb*32 + 16*(lane>>4) + e) * C + nt*16 + (lane&15)].
// ---------------------------------------------------------------------------
__global__ __launch_bounds__(256) void k0_prep(const float* __restrict__ WA,
                                               const float* __restrict__ Key,
                                               const float* __restrict__ Mem,
                                               u16* __restrict__ fWA,
                                               u16* __restrict__ fKey,
                                               u16* __restrict__ fMem) {
  int i = blockIdx.x * 256 + threadIdx.x;
  auto conv = [](const float* src, u16* dst, int idx, int ktiles, int C) {
    int e  = idx & 15;
    int L  = (idx >> 4) & 31;
    int t  = idx >> 9;
    int kb = t % ktiles;
    int nt = t / ktiles;
    int k  = kb * 32 + 16 * (L >> 4) + e;
    int c  = nt * 16 + (L & 15);
    dst[idx] = f2bf(src[(size_t)k * C + c]);
  };
  if (i < 65536)            conv(WA,  fWA,  i,          8, 256);
  else if (i < 65536+8192)  conv(Key, fKey, i - 65536,  8, 32);
  else if (i < 81920)       conv(Mem, fMem, i - 73728,  1, 256);
}

// ---------------------------------------------------------------------------
// k1: one block per batch element. 256 threads = 8 waves.
// ---------------------------------------------------------------------------
__global__ __launch_bounds__(256) void k1_attkey(const int* __restrict__ input_u,
                                                 const int* __restrict__ input_uf,
                                                 const float* __restrict__ uidW,
                                                 const u16* __restrict__ fKey,
                                                 float* __restrict__ att_key) {
  __shared__ __align__(16) u16 s_cross[NFF * DD];   // cross, bf16, 32 KB
  __shared__ float s_uidn[DD];
  __shared__ float s_red[8];

  const int b = blockIdx.x, t = threadIdx.x;
  const int lane = t & 31, w = t >> 5;
  const int hL = lane >> 4, nL = lane & 15;

  // uid row + L2 norm -> s_uidn
  int u = input_u[b];
  float x  = uidW[(size_t)u * DD + t];
  float ss = blockReduceSum(x * x, s_red);
  s_uidn[t] = x * (1.0f / fmaxf(sqrtf(ss), 1e-12f));
  __syncthreads();

  // fe rows: one wave per row, 8 rows per wave; mask-by-multiply (index
  // USER_NUM is a real row of uidW, so loads are always in-bounds & uniform).
  const float4* uidn4 = (const float4*)s_uidn;
  for (int it = 0; it < 8; ++it) {
    int f = w + 8 * it;
    int idx = input_uf[b * NFF + f];
    float maskf = (idx != USER_NUMC) ? 1.f : 0.f;
    const float4* src = (const float4*)(uidW + (size_t)idx * DD);
    if (it < 7) {
      int idx2 = input_uf[b * NFF + f + 8];
      __builtin_prefetch(uidW + (size_t)idx2 * DD + lane * 4, 0, 1);
    }
    float4 a0 = src[lane];        // d = lane*4 .. +3
    float4 a1 = src[lane + 32];   // d = 128 + lane*4 .. +3
    a0.x *= maskf; a0.y *= maskf; a0.z *= maskf; a0.w *= maskf;
    a1.x *= maskf; a1.y *= maskf; a1.z *= maskf; a1.w *= maskf;
    float sq = a0.x*a0.x + a0.y*a0.y + a0.z*a0.z + a0.w*a0.w
             + a1.x*a1.x + a1.y*a1.y + a1.z*a1.z + a1.w*a1.w;
#pragma unroll
    for (int off = 16; off > 0; off >>= 1) sq += __shfl_xor(sq, off, 32);
    float invf = 1.0f / fmaxf(sqrtf(sq), 1e-12f);
    // cross = fe_n * uid_n = fe * invf * uid_n, stored bf16
    float4 u0 = uidn4[lane], u1 = uidn4[lane + 32];
    uint2 p0, p1;
    p0.x = pk2(a0.x * u0.x * invf, a0.y * u0.y * invf);
    p0.y = pk2(a0.z * u0.z * invf, a0.w * u0.w * invf);
    p1.x = pk2(a1.x * u1.x * invf, a1.y * u1.y * invf);
    p1.y = pk2(a1.z * u1.z * invf, a1.w * u1.w * invf);
    ((uint2*)s_cross)[f * (DD/4) + lane]      = p0;
    ((uint2*)s_cross)[f * (DD/4) + 32 + lane] = p1;
  }
  __syncthreads();

  // GEMM1: cross[64,256] @ Key[256,32] -> att_key[64,32]
  const int mt = w >> 1, nt = w & 1;
  const int row = mt * 16 + nL;
  v8f acc = {0.f,0.f,0.f,0.f,0.f,0.f,0.f,0.f};
  for (int kb = 0; kb < 8; ++kb) {
    const uint4* pa = (const uint4*)(s_cross + row * DD + kb * 32 + 16 * hL);
    union { uint4 q[2]; v16bf v; } ua; ua.q[0] = pa[0]; ua.q[1] = pa[1];
    const uint4* pb = (const uint4*)(fKey + (((size_t)(nt * 8 + kb) * 32 + lane) * 16));
    union { uint4 q[2]; v16bf v; } ub; ub.q[0] = pb[0]; ub.q[1] = pb[1];
    acc = __builtin_amdgcn_wmma_f32_16x16x32_bf16(false, ua.v, false, ub.v,
                                                  (short)0, acc, false, false);
  }
  float* ak = att_key + (size_t)b * NFF * MM;
#pragma unroll
  for (int r = 0; r < 8; ++r) {
    int f = mt * 16 + 8 * hL + r;
    int m = nt * 16 + nL;
    ak[f * MM + m] = acc[r];
  }
}

// ---------------------------------------------------------------------------
// k2: per (f,m) online max / sum-exp over the batch axis. 64 blocks (one per f).
// ---------------------------------------------------------------------------
__global__ __launch_bounds__(256) void k2_stats(const float* __restrict__ att_key,
                                                float* __restrict__ wmax,
                                                float* __restrict__ wsum) {
  __shared__ float sm[256], ssv[256];
  const int f = blockIdx.x, t = threadIdx.x;
  const int m = t & 31, g = t >> 5;
  float M = -3.0e38f, S = 0.f;
  for (int b = g; b < BB; b += 8) {
    float x = att_key[((size_t)b * NFF + f) * MM + m];
    float nm = fmaxf(M, x);
    S = S * __expf(M - nm) + __expf(x - nm);
    M = nm;
  }
  sm[t] = M; ssv[t] = S;
  __syncthreads();
  if (t < 32) {
    float MM2 = sm[t], SS = ssv[t];
#pragma unroll
    for (int gg = 1; gg < 8; ++gg) {
      float m2 = sm[gg * 32 + t], s2 = ssv[gg * 32 + t];
      float nm = fmaxf(MM2, m2);
      SS = SS * __expf(MM2 - nm) + s2 * __expf(m2 - nm);
      MM2 = nm;
    }
    wmax[f * MM + t] = MM2;
    wsum[f * MM + t] = SS;
  }
}

// ---------------------------------------------------------------------------
// k3: one block per batch element; full fused tail.
// ---------------------------------------------------------------------------
__global__ __launch_bounds__(256) void k3_main(const int* __restrict__ input_u,
                                               const int* __restrict__ input_i,
                                               const int* __restrict__ input_uf,
                                               const float* __restrict__ uidW,
                                               const float* __restrict__ iidW,
                                               const float* __restrict__ i_bias,
                                               const float* __restrict__ BA,
                                               const float* __restrict__ U_omega,
                                               const float* __restrict__ att_exp1,
                                               const float* __restrict__ att_key,
                                               const float* __restrict__ wmax,
                                               const float* __restrict__ wsum,
                                               const u16* __restrict__ fMem,
                                               const u16* __restrict__ fWA,
                                               float* __restrict__ out) {
  __shared__ __align__(16) u16 s_fe[NFF * DD];   // fe, then f2 (bf16)   32 KB
  __shared__ __align__(16) float s_ak[NFF * MM]; // staged att_key        8 KB
  __shared__ __align__(16) u16 s_att[NFF * MM];  // att_mem (bf16)        4 KB
  __shared__ float s_uid[DD];
  __shared__ float s_j[NFF];
  __shared__ float s_w[NFF];
  __shared__ float s_mask[NFF];
  __shared__ float s_jp[NFF * 2];
  __shared__ float s_red[8];

  const int b = blockIdx.x, t = threadIdx.x;
  const int lane = t & 31, w = t >> 5;
  const int hL = lane >> 4, nL = lane & 15;

  const float* ak = att_key + (size_t)b * NFF * MM;

  // ---- stage att_key[b] (8 KB contiguous) into LDS via TDM / async DMA ----
#if HAVE_TDM
  if (w == 0) {
    unsigned long long ga = (unsigned long long)(const void*)ak;
    unsigned lds_off = (unsigned)(unsigned long long)(LAS char*)(void*)s_ak;
    v4u g0;
    g0[0] = 1u;                                     // count=1, user descriptor
    g0[1] = lds_off;                                // lds_addr
    g0[2] = (unsigned)(ga & 0xffffffffu);           // global_addr[31:0]
    g0[3] = (unsigned)((ga >> 32) & 0x01ffffffu) | (2u << 30); // addr[56:32], type=2
    v8i g1;
    g1[0] = 0x20000;                  // data_size=2 (4B), no mask/iterate/pad
    g1[1] = (int)((2048u & 0xffffu) << 16);         // tensor_dim0 = 2048 (lo)
    g1[2] = (int)(1u << 16);                        // tensor_dim1 = 1
    g1[3] = (int)(2048u << 16);                     // tile_dim0 = 2048
    g1[4] = 0;                                      // tile_dim1/2 unused (1-D)
    g1[5] = 2048;                                   // tensor_dim0_stride
    g1[6] = 0;
    g1[7] = 0;
    v4i gz4 = {0, 0, 0, 0};
    v8i gz8 = {0, 0, 0, 0, 0, 0, 0, 0};
    // amdgpu-toolchain (clang-23) 6-arg form: (g0, g1, g2, g3, g4, cpol)
    __builtin_amdgcn_tensor_load_to_lds(g0, g1, gz4, gz4, gz8, 0);
  }
#elif HAVE_ASYNC
  __builtin_amdgcn_global_load_async_to_lds_b128(
      (GAS void*)(unsigned long long)(const void*)(ak + t * 4),
      (LAS void*)((char*)(void*)s_ak + t * 16), 0, 0);
  __builtin_amdgcn_global_load_async_to_lds_b128(
      (GAS void*)(unsigned long long)(const void*)(ak + 1024 + t * 4),
      (LAS void*)((char*)(void*)s_ak + 4096 + t * 16), 0, 0);
#else
#pragma unroll
  for (int i = 0; i < 8; ++i) s_ak[t + 256 * i] = ak[t + 256 * i];
#endif

  int u = input_u[b];
  s_uid[t] = uidW[(size_t)u * DD + t];
  if (t < NFF) s_mask[t] = (input_uf[b * NFF + t] != USER_NUMC) ? 1.f : 0.f;

  // regather masked fe rows into LDS (bf16), b128 loads, mask-by-multiply
  for (int it = 0; it < 8; ++it) {
    int f = w + 8 * it;
    int idx = input_uf[b * NFF + f];
    float maskf = (idx != USER_NUMC) ? 1.f : 0.f;
    const float4* src = (const float4*)(uidW + (size_t)idx * DD);
    if (it < 7) {
      int idx2 = input_uf[b * NFF + f + 8];
      __builtin_prefetch(uidW + (size_t)idx2 * DD + lane * 4, 0, 1);
    }
    float4 a0 = src[lane];
    float4 a1 = src[lane + 32];
    uint2 p0, p1;
    p0.x = pk2(a0.x * maskf, a0.y * maskf);
    p0.y = pk2(a0.z * maskf, a0.w * maskf);
    p1.x = pk2(a1.x * maskf, a1.y * maskf);
    p1.y = pk2(a1.z * maskf, a1.w * maskf);
    ((uint2*)s_fe)[f * (DD/4) + lane]      = p0;
    ((uint2*)s_fe)[f * (DD/4) + 32 + lane] = p1;
  }

  // wait for the staged att_key, then barrier
#if HAVE_TDM
  if (w == 0) __builtin_amdgcn_s_wait_tensorcnt(0);
#elif HAVE_ASYNC
  __builtin_amdgcn_s_wait_asynccnt(0);
#endif
  __syncthreads();

  // att_mem = mask * exp(att_key - max_b) / sumexp_b   (bf16 in LDS)
#pragma unroll
  for (int i = 0; i < 8; ++i) {
    int idx = t + 256 * i;
    int f = idx >> 5, m = idx & 31;
    s_att[idx] = f2bf(s_mask[f] * __expf(s_ak[idx] - wmax[f * MM + m]) / wsum[f * MM + m]);
  }
  __syncthreads();

  // GEMM3: f1 = att_mem[64,32] @ Mem[32,256]; fused f2 = f1*fe -> s_fe
  for (int i = 0; i < 8; ++i) {
    int tid = w + 8 * i;
    int mt = tid >> 4, nt = tid & 15;
    int fr = mt * 16 + nL;
    const uint4* pa = (const uint4*)(s_att + fr * MM + 16 * hL);
    union { uint4 q[2]; v16bf v; } ua; ua.q[0] = pa[0]; ua.q[1] = pa[1];
    const uint4* pb = (const uint4*)(fMem + (((size_t)nt * 32 + lane) * 16));
    union { uint4 q[2]; v16bf v; } ub; ub.q[0] = pb[0]; ub.q[1] = pb[1];
    v8f acc = {0.f,0.f,0.f,0.f,0.f,0.f,0.f,0.f};
    acc = __builtin_amdgcn_wmma_f32_16x16x32_bf16(false, ua.v, false, ub.v,
                                                  (short)0, acc, false, false);
#pragma unroll
    for (int r = 0; r < 8; ++r) {
      int f = mt * 16 + 8 * hL + r;
      int d = nt * 16 + nL;
      float f2v = acc[r] * bf2f(s_fe[f * DD + d]);
      s_fe[f * DD + d] = f2bf(f2v);          // 1:1 overwrite, no cross-tile aliasing
    }
  }
  __syncthreads();

  // GEMM2: h = relu(f2[64,256] @ WA[256,256] + BA); fused j-logit = h @ U_omega
  {
    int mt = w & 3, ntg = w >> 2;
    float jacc[8];
#pragma unroll
    for (int r = 0; r < 8; ++r) jacc[r] = 0.f;
    for (int i = 0; i < 8; ++i) {
      int nt = ntg + 2 * i;
      v8f acc = {0.f,0.f,0.f,0.f,0.f,0.f,0.f,0.f};
      for (int kb = 0; kb < 8; ++kb) {
        const uint4* pa = (const uint4*)(s_fe + (mt * 16 + nL) * DD + kb * 32 + 16 * hL);
        union { uint4 q[2]; v16bf v; } ua; ua.q[0] = pa[0]; ua.q[1] = pa[1];
        const uint4* pb = (const uint4*)(fWA + (((size_t)(nt * 8 + kb) * 32 + lane) * 16));
        union { uint4 q[2]; v16bf v; } ub; ub.q[0] = pb[0]; ub.q[1] = pb[1];
        acc = __builtin_amdgcn_wmma_f32_16x16x32_bf16(false, ua.v, false, ub.v,
                                                      (short)0, acc, false, false);
      }
      int acol = nt * 16 + nL;
      float ba = BA[acol], uo = U_omega[acol];
#pragma unroll
      for (int r = 0; r < 8; ++r) jacc[r] += fmaxf(acc[r] + ba, 0.f) * uo;
    }
    // reduce across the 16 N-lanes; deterministic per-wave partial store
#pragma unroll
    for (int r = 0; r < 8; ++r) {
      float v = jacc[r];
#pragma unroll
      for (int off = 1; off < 16; off <<= 1) v += __shfl_xor(v, off, 32);
      if (nL == 0) s_jp[(mt * 16 + 8 * hL + r) * 2 + ntg] = v;
    }
  }
  __syncthreads();

  // j, sum_j, w
  if (t < NFF) s_j[t] = s_mask[t] * __expf(s_jp[2 * t] + s_jp[2 * t + 1]);
  __syncthreads();
  if (t < 32) {
    float v = s_j[t] + s_j[t + 32];
#pragma unroll
    for (int off = 16; off > 0; off >>= 1) v += __shfl_xor(v, off, 32);
    if (t == 0) s_red[0] = v;
  }
  __syncthreads();
  float sumj = s_red[0];
  __syncthreads();
  if (t < NFF) s_w[t] = s_j[t] / (sumj + 1e-8f);
  __syncthreads();

  // friend, user, 2-way attention, score (thread t owns dimension d = t)
  float fr = 0.f;
  for (int f = 0; f < NFF; ++f) fr += s_w[f] * bf2f(s_fe[f * DD + t]);
  float uv   = s_uid[t];
  float user = uv + fr;
  float p0 = uv * att_exp1[t * 2 + 0] + user * att_exp1[(DD + t) * 2 + 0];
  float p1 = uv * att_exp1[t * 2 + 1] + user * att_exp1[(DD + t) * 2 + 1];
  float l0 = blockReduceSum(p0, s_red);
  float l1 = blockReduceSum(p1, s_red);
  float mx = fmaxf(l0, l1);
  float e0 = __expf(l0 - mx), e1 = __expf(l1 - mx);
  float a0 = e0 / (e0 + e1), a1 = e1 / (e0 + e1);
  float user_f = uv * a0 + user * a1;
  int ii = input_i[b];
  float sc = blockReduceSum(user_f * iidW[(size_t)ii * DD + t], s_red);
  if (t == 0) out[b] = sc + i_bias[ii];
}

// ---------------------------------------------------------------------------
extern "C" void kernel_launch(void* const* d_in, const int* in_sizes, int n_in,
                              void* d_out, int out_size, void* d_ws, size_t ws_size,
                              hipStream_t stream) {
  const int*   input_u  = (const int*)  d_in[0];
  const int*   input_i  = (const int*)  d_in[1];
  const int*   input_uf = (const int*)  d_in[2];
  const float* uidW     = (const float*)d_in[3];
  const float* iidW     = (const float*)d_in[4];
  const float* i_bias   = (const float*)d_in[5];
  const float* Key      = (const float*)d_in[6];
  const float* Mem      = (const float*)d_in[7];
  const float* WA       = (const float*)d_in[8];
  const float* BA       = (const float*)d_in[9];
  const float* U_omega  = (const float*)d_in[10];
  const float* att_exp1 = (const float*)d_in[11];
  float* out = (float*)d_out;

  char* ws = (char*)d_ws;
  float* att_key = (float*)(ws + OFF_ATT);
  float* wmax    = (float*)(ws + OFF_MAX);
  float* wsum    = (float*)(ws + OFF_SUM);
  u16*   fKey    = (u16*)  (ws + OFF_FKEY);
  u16*   fMem    = (u16*)  (ws + OFF_FMEM);
  u16*   fWA     = (u16*)  (ws + OFF_FWA);

  k0_prep<<<320, 256, 0, stream>>>(WA, Key, Mem, fWA, fKey, fMem);
  k1_attkey<<<BB, 256, 0, stream>>>(input_u, input_uf, uidW, fKey, att_key);
  k2_stats<<<NFF, 256, 0, stream>>>(att_key, wmax, wsum);
  k3_main<<<BB, 256, 0, stream>>>(input_u, input_i, input_uf, uidW, iidW, i_bias,
                                  BA, U_omega, att_exp1, att_key, wmax, wsum,
                                  fMem, fWA, out);
}
